// DiM_90452011254020
// MI455X (gfx1250) — matrix operand order repaired and checked by reading, not compiled, not run on hardware
//
#include <hip/hip_runtime.h>
#include <math.h>

#define D_MODEL 768
#define D_INNER 1536
#define NSTATE  64
#define DTR     48
#define LSEQ    1024
#define NBATCH  4
#define MTOT    (NBATCH*LSEQ)      // 4096 rows
#define DBL_N   (DTR + 2*NSTATE)   // 176

typedef __attribute__((ext_vector_type(16))) _Float16 v16h;
typedef __attribute__((ext_vector_type(8)))  _Float16 v8h;
typedef __attribute__((ext_vector_type(4)))  _Float16 h4;
typedef __attribute__((ext_vector_type(8)))  float    v8f;

enum { EP_NONE = 0, EP_GELU = 1, EP_SOFTPLUS = 2, EP_GATE_RES = 3 };

__device__ __forceinline__ float sigmoidf_(float x) { return 1.f / (1.f + __expf(-x)); }
__device__ __forceinline__ float siluf_(float x)    { return x * sigmoidf_(x); }

// ---------------------------------------------------------------------------
// WMMA GEMM: C[M,N] = epilogue(A[M,K] @ B[K,N] + bias)
// 256 threads = 8 waves; block tile 128x128, BK=32; wave tile 32x64 (2x4 WMMA)
// Software pipeline: float4 global loads for tile kt+1 are issued before the
// WMMA compute on tile kt, and written to LDS at the top of the next iter.
// Block-uniform fast path for interior tiles (no exec-mask guards).
// Requires: lda, ldb, K, N multiples of 4 (true for every call below).
// ---------------------------------------------------------------------------
__global__ __launch_bounds__(256)
void gemm_f16_wmma(const float* __restrict__ A, const float* __restrict__ B,
                   const float* __restrict__ bias, float* __restrict__ C,
                   int M, int N, int K, int lda, int ldb, int ldc,
                   int mode, const float* __restrict__ gate, int gstride)
{
    __shared__ _Float16 As[128][40];   // 128 rows x 32 K (+pad), 80B pitch
    __shared__ _Float16 Bs[32][136];   // 32 K rows x 128 N (+pad), 272B pitch

    const int m0   = blockIdx.y * 128;
    const int n0   = blockIdx.x * 128;
    const int tid  = threadIdx.x;
    const int lane = tid & 31;
    const int wave = tid >> 5;
    const int wm   = wave >> 1;       // 0..3 : 32-row slab
    const int wn   = wave & 1;        // 0..1 : 64-col slab
    const int mrow = lane & 15;
    const int hlf  = lane >> 4;

    v8f acc[2][4];
#pragma unroll
    for (int mi = 0; mi < 2; ++mi)
#pragma unroll
        for (int ni = 0; ni < 4; ++ni)
#pragma unroll
            for (int r = 0; r < 8; ++r) acc[mi][ni][r] = 0.f;

    const float4 z4 = make_float4(0.f, 0.f, 0.f, 0.f);
    float4 ar[4], br[4];

    // Per-thread staging coordinates (A: 128 rows x 8 float4; B: 32 rows x 32 float4)
    const int aR = tid >> 3, aC = (tid & 7) << 2;    // +32 rows per i
    const int bR = tid >> 5, bC = (tid & 31) << 2;   // +8 rows per i
    const float* aBase = A + (size_t)(m0 + aR) * lda + aC;
    const float* bBase = B + (size_t)(bR) * ldb + (n0 + bC);
    const size_t aStep = (size_t)32 * lda;
    const size_t bStep = (size_t)8 * ldb;
    const bool fullMN = (m0 + 128 <= M) && (n0 + 128 <= N);

    // ---- issue float4 global loads for K-tile at k0 into registers ----
    auto load_tiles = [&](int k0) {
        if (fullMN && (k0 + 32 <= K)) {            // interior tile: branch-free
#pragma unroll
            for (int i = 0; i < 4; ++i)
                ar[i] = *(const float4*)(aBase + (size_t)i * aStep + k0);
#pragma unroll
            for (int i = 0; i < 4; ++i)
                br[i] = *(const float4*)(bBase + (size_t)i * bStep + (size_t)k0 * ldb);
        } else {                                   // edge tile: guarded
#pragma unroll
            for (int i = 0; i < 4; ++i) {
                int gm = m0 + aR + 32 * i, gk = k0 + aC;
                ar[i] = (gm < M && gk < K) ? *(const float4*)&A[(size_t)gm * lda + gk] : z4;
            }
#pragma unroll
            for (int i = 0; i < 4; ++i) {
                int gk = k0 + bR + 8 * i, gn = n0 + bC;
                br[i] = (gk < K && gn < N) ? *(const float4*)&B[(size_t)gk * ldb + gn] : z4;
            }
        }
    };
    // ---- convert + store staged registers into LDS ----
    auto store_tiles = [&]() {
#pragma unroll
        for (int i = 0; i < 4; ++i) {
            h4 hv = { (_Float16)ar[i].x, (_Float16)ar[i].y,
                      (_Float16)ar[i].z, (_Float16)ar[i].w };
            *(h4*)&As[aR + 32 * i][aC] = hv;
        }
#pragma unroll
        for (int i = 0; i < 4; ++i) {
            h4 hv = { (_Float16)br[i].x, (_Float16)br[i].y,
                      (_Float16)br[i].z, (_Float16)br[i].w };
            *(h4*)&Bs[bR + 8 * i][bC] = hv;
        }
    };

    const int ktiles = (K + 31) >> 5;
    load_tiles(0);
    for (int kt = 0; kt < ktiles; ++kt) {
        store_tiles();                         // waits on in-flight loads
        __syncthreads();
        if (kt + 1 < ktiles) load_tiles((kt + 1) << 5);  // overlap with compute

        // ---- fragment loads per documented CDNA5 VGPR layouts ----
        v16h afrag[2], bfrag[4];
#pragma unroll
        for (int mi = 0; mi < 2; ++mi) {
            const _Float16* ap = &As[wm * 32 + mi * 16 + mrow][0];
            v8h lo = *(const v8h*)(ap + 8 * hlf);        // K = 8*half .. +7
            v8h hi = *(const v8h*)(ap + 16 + 8 * hlf);   // K = 16+8*half .. +7
            v16h t;
#pragma unroll
            for (int e = 0; e < 8; ++e) { t[e] = lo[e]; t[e + 8] = hi[e]; }
            afrag[mi] = t;
        }
#pragma unroll
        for (int ni = 0; ni < 4; ++ni) {
            const int bcol = wn * 64 + ni * 16 + mrow;         // N = lane%16
            v16h t;
#pragma unroll
            for (int e = 0; e < 8; ++e) { t[e] = Bs[8 * hlf + e][bcol]; t[e + 8] = Bs[16 + 8 * hlf + e][bcol]; }
            bfrag[ni] = t;
        }
#pragma unroll
        for (int mi = 0; mi < 2; ++mi)
#pragma unroll
            for (int ni = 0; ni < 4; ++ni)
                acc[mi][ni] = __builtin_amdgcn_wmma_f32_16x16x32_f16(
                    false, afrag[mi], false, bfrag[ni],
                    (short)0, acc[mi][ni], false, false);
        __syncthreads();
    }

    // ---- epilogue: D layout is M = r + 8*(lane/16), N = lane%16 ----
#pragma unroll
    for (int mi = 0; mi < 2; ++mi)
#pragma unroll
        for (int ni = 0; ni < 4; ++ni) {
            int col = n0 + wn * 64 + ni * 16 + mrow;
            if (col >= N) continue;
            float bv = bias ? bias[col] : 0.f;
#pragma unroll
            for (int r = 0; r < 8; ++r) {
                int row = m0 + wm * 32 + mi * 16 + hlf * 8 + r;
                if (row >= M) continue;
                float v = acc[mi][ni][r] + bv;
                if (mode == EP_GELU) {
                    v = 0.5f * v * (1.f + tanhf(0.7978845608f * (v + 0.044715f * v * v * v)));
                } else if (mode == EP_SOFTPLUS) {
                    v = (v > 20.f) ? v : log1pf(__expf(v));
                }
                size_t off = (size_t)row * ldc + col;
                if (mode == EP_GATE_RES) {
                    int b = row / LSEQ;
                    C[off] += gate[(size_t)b * gstride + col] * v;
                } else {
                    C[off] = v;
                }
            }
        }
}

// ---------------------------------------------------------------------------
// x-embedder: h = concat(x, y_noised) @ xe_w + xe_b + pos
// ---------------------------------------------------------------------------
__global__ void embed_kernel(const float* __restrict__ x, const float* __restrict__ yn,
                             const float* __restrict__ xe_w, const float* __restrict__ xe_b,
                             const float* __restrict__ pos, float* __restrict__ h)
{
    int idx = blockIdx.x * blockDim.x + threadIdx.x;
    if (idx >= MTOT * D_MODEL) return;
    int m = idx / D_MODEL, n = idx % D_MODEL;
    int l = m % LSEQ;
    float acc = xe_b[n] + pos[(size_t)l * D_MODEL + n];
    const float* xr = x  + (size_t)m * 16;
    const float* yr = yn + (size_t)m * 16;
#pragma unroll
    for (int j = 0; j < 16; ++j) {
        acc += xr[j] * xe_w[(size_t)j * D_MODEL + n];
        acc += yr[j] * xe_w[(size_t)(16 + j) * D_MODEL + n];
    }
    h[idx] = acc;
}

// ---------------------------------------------------------------------------
// timestep sinusoidal embedding (B x 256)
// ---------------------------------------------------------------------------
__global__ void tfreq_kernel(const float* __restrict__ t, float* __restrict__ emb)
{
    int b = blockIdx.x, i = threadIdx.x; // 256 threads
    int j = (i < 128) ? i : i - 128;
    float f = __expf(-logf(10000.f) * (float)j / 128.f);
    float a = t[b] * f;
    emb[b * 256 + i] = (i < 128) ? cosf(a) : sinf(a);
}

// ---------------------------------------------------------------------------
// small GEMV over batch rows: out[b,n] = act(X[b,:] @ W[:,n] + bias[n])
// ---------------------------------------------------------------------------
__global__ void gemv_kernel(const float* __restrict__ X, const float* __restrict__ W,
                            const float* __restrict__ bias, float* __restrict__ out,
                            int K, int N, int act)
{
    int idx = blockIdx.x * blockDim.x + threadIdx.x;
    if (idx >= NBATCH * N) return;
    int b = idx / N, n = idx % N;
    float acc = bias ? bias[n] : 0.f;
    for (int k = 0; k < K; ++k) acc += X[(size_t)b * K + k] * W[(size_t)k * N + n];
    if (act == 1) acc = siluf_(acc);
    out[idx] = acc;
}

// ---------------------------------------------------------------------------
// LayerNorm (eps 1e-6, no affine) + adaLN modulate
// ---------------------------------------------------------------------------
__global__ __launch_bounds__(256)
void lnmod_kernel(const float* __restrict__ h, float* __restrict__ out,
                  const float* __restrict__ shift, const float* __restrict__ scale,
                  int modStride)
{
    __shared__ float red[256];
    int row = blockIdx.x;
    int b = row / LSEQ;
    const float* hr = h + (size_t)row * D_MODEL;
    int tid = threadIdx.x;

    float s = 0.f;
    for (int j = tid; j < D_MODEL; j += 256) s += hr[j];
    red[tid] = s; __syncthreads();
    for (int st = 128; st > 0; st >>= 1) { if (tid < st) red[tid] += red[tid + st]; __syncthreads(); }
    float mu = red[0] / (float)D_MODEL;
    __syncthreads();

    float v = 0.f;
    for (int j = tid; j < D_MODEL; j += 256) { float d = hr[j] - mu; v += d * d; }
    red[tid] = v; __syncthreads();
    for (int st = 128; st > 0; st >>= 1) { if (tid < st) red[tid] += red[tid + st]; __syncthreads(); }
    float rstd = rsqrtf(red[0] / (float)D_MODEL + 1e-6f);

    const float* sh = shift + (size_t)b * modStride;
    const float* sc = scale + (size_t)b * modStride;
    float* orow = out + (size_t)row * D_MODEL;
    for (int j = tid; j < D_MODEL; j += 256)
        orow[j] = (hr[j] - mu) * rstd * (1.f + sc[j]) + sh[j];
}

// ---------------------------------------------------------------------------
// depthwise causal conv (K=4) over L + SiLU; input = xz[:, :, 0:D_INNER]
// ---------------------------------------------------------------------------
__global__ void conv_silu_kernel(const float* __restrict__ xz, const float* __restrict__ cw,
                                 const float* __restrict__ cb, float* __restrict__ u)
{
    int idx = blockIdx.x * blockDim.x + threadIdx.x;
    if (idx >= MTOT * D_INNER) return;
    int d = idx % D_INNER, m = idx / D_INNER;
    int l = m % LSEQ, b = m / LSEQ;
    float acc = cb[d];
#pragma unroll
    for (int k = 0; k < 4; ++k) {
        int ls = l - 3 + k;
        if (ls >= 0)
            acc += xz[((size_t)(b * LSEQ + ls)) * (2 * D_INNER) + d] * cw[d * 4 + k];
    }
    u[idx] = siluf_(acc);
}

// ---------------------------------------------------------------------------
// y *= silu(z), z = xz[:, :, D_INNER:]
// ---------------------------------------------------------------------------
__global__ void gatez_kernel(const float* __restrict__ xz, float* __restrict__ y)
{
    int idx = blockIdx.x * blockDim.x + threadIdx.x;
    if (idx >= MTOT * D_INNER) return;
    int d = idx % D_INNER, m = idx / D_INNER;
    float z = xz[(size_t)m * (2 * D_INNER) + D_INNER + d];
    y[idx] *= siluf_(z);
}

// ---------------------------------------------------------------------------
// Selective scan: one thread per (b, d) channel, 64-wide state in registers,
// B_t/C_t broadcast through LDS each timestep.
// grid = (D_INNER/256, NBATCH), block = 256
// ---------------------------------------------------------------------------
__global__ __launch_bounds__(256)
void scan_kernel(const float* __restrict__ u, const float* __restrict__ dt,
                 const float* __restrict__ dbl, const float* __restrict__ A_log,
                 const float* __restrict__ Dp, float* __restrict__ y)
{
    __shared__ float Bs[NSTATE], Cs[NSTATE];
    int b = blockIdx.y;
    int d = blockIdx.x * 256 + threadIdx.x;

    float Areg[NSTATE], h[NSTATE];
#pragma unroll
    for (int n = 0; n < NSTATE; ++n) {
        Areg[n] = -__expf(A_log[(size_t)d * NSTATE + n]);
        h[n] = 0.f;
    }
    float Dv = Dp[d];

    for (int t = 0; t < LSEQ; ++t) {
        __syncthreads();
        if (threadIdx.x < 2 * NSTATE) {
            float v = dbl[(size_t)(b * LSEQ + t) * DBL_N + DTR + threadIdx.x];
            if (threadIdx.x < NSTATE) Bs[threadIdx.x] = v;
            else                      Cs[threadIdx.x - NSTATE] = v;
        }
        __syncthreads();
        size_t idx = (size_t)(b * LSEQ + t) * D_INNER + d;
        float dtv = dt[idx], uv = u[idx];
        float du = dtv * uv, acc = 0.f;
#pragma unroll
        for (int n = 0; n < NSTATE; ++n) {
            h[n] = h[n] * __expf(dtv * Areg[n]) + du * Bs[n];
            acc += h[n] * Cs[n];
        }
        y[idx] = acc + Dv * uv;
    }
}

// ---------------------------------------------------------------------------
// host orchestration
// ---------------------------------------------------------------------------
static inline void launch_gemm(const float* A, const float* B, const float* bias, float* C,
                               int M, int N, int K, int lda, int ldb, int ldc,
                               int mode, const float* gate, int gstride, hipStream_t s)
{
    dim3 grid((N + 127) / 128, (M + 127) / 128);
    gemm_f16_wmma<<<grid, 256, 0, s>>>(A, B, bias, C, M, N, K, lda, ldb, ldc, mode, gate, gstride);
}

extern "C" void kernel_launch(void* const* d_in, const int* in_sizes, int n_in,
                              void* d_out, int out_size, void* d_ws, size_t ws_size,
                              hipStream_t stream)
{
    const float* x      = (const float*)d_in[0];
    const float* yn     = (const float*)d_in[1];
    const float* tv     = (const float*)d_in[2];
    const float* xe_w   = (const float*)d_in[3];
    const float* xe_b   = (const float*)d_in[4];
    const float* pos    = (const float*)d_in[5];
    const float* te_w1  = (const float*)d_in[6];
    const float* te_b1  = (const float*)d_in[7];
    const float* te_w2  = (const float*)d_in[8];
    const float* te_b2  = (const float*)d_in[9];
    const float* ada_w  = (const float*)d_in[10];
    const float* ada_b  = (const float*)d_in[11];
    const float* in_w   = (const float*)d_in[12];
    const float* in_b   = (const float*)d_in[13];
    const float* conv_w = (const float*)d_in[14];
    const float* conv_b = (const float*)d_in[15];
    const float* xp_w   = (const float*)d_in[16];
    const float* dtp_w  = (const float*)d_in[17];
    const float* dtp_b  = (const float*)d_in[18];
    const float* A_log  = (const float*)d_in[19];
    const float* Dp     = (const float*)d_in[20];
    const float* out_w  = (const float*)d_in[21];
    const float* out_b  = (const float*)d_in[22];
    const float* fc1_w  = (const float*)d_in[23];
    const float* fc1_b  = (const float*)d_in[24];
    const float* fc2_w  = (const float*)d_in[25];
    const float* fc2_b  = (const float*)d_in[26];
    const float* flaw   = (const float*)d_in[27];
    const float* flab   = (const float*)d_in[28];
    const float* fl_w   = (const float*)d_in[29];
    const float* fl_b   = (const float*)d_in[30];

    float* ws = (float*)d_ws;
    size_t o = 0;
    float* hbuf  = ws + o; o += (size_t)MTOT * D_MODEL;        // 4096x768
    float* xmod  = ws + o; o += (size_t)MTOT * D_MODEL;        // 4096x768
    float* xz    = ws + o; o += (size_t)MTOT * 2 * D_INNER;    // 4096x3072 (also MLP hidden)
    float* ubuf  = ws + o; o += (size_t)MTOT * D_INNER;        // 4096x1536
    float* dblb  = ws + o; o += (size_t)MTOT * DBL_N;          // 4096x176
    float* dtb   = ws + o; o += (size_t)MTOT * D_INNER;        // 4096x1536
    float* ybuf  = ws + o; o += (size_t)MTOT * D_INNER;        // 4096x1536
    float* emb   = ws + o; o += (size_t)NBATCH * 256;
    float* h1    = ws + o; o += (size_t)NBATCH * D_MODEL;
    float* scb   = ws + o; o += (size_t)NBATCH * D_MODEL;
    float* mods  = ws + o; o += (size_t)NBATCH * 6 * D_MODEL;
    float* fmods = ws + o; o += (size_t)NBATCH * 2 * D_MODEL;

    const int TPB = 256;

    // h = concat(x, y) @ xe_w + b + pos
    embed_kernel<<<(MTOT * D_MODEL + TPB - 1) / TPB, TPB, 0, stream>>>(x, yn, xe_w, xe_b, pos, hbuf);

    // conditioning: sc = silu(silu(emb@w1+b1)@w2+b2)
    tfreq_kernel<<<NBATCH, 256, 0, stream>>>(tv, emb);
    gemv_kernel<<<(NBATCH * D_MODEL + TPB - 1) / TPB, TPB, 0, stream>>>(emb, te_w1, te_b1, h1, 256, D_MODEL, 1);
    gemv_kernel<<<(NBATCH * D_MODEL + TPB - 1) / TPB, TPB, 0, stream>>>(h1, te_w2, te_b2, scb, D_MODEL, D_MODEL, 1);

    for (int i = 0; i < 4; ++i) {
        const float* ada_wi = ada_w + (size_t)i * D_MODEL * 6 * D_MODEL;
        const float* ada_bi = ada_b + (size_t)i * 6 * D_MODEL;
        gemv_kernel<<<(NBATCH * 6 * D_MODEL + TPB - 1) / TPB, TPB, 0, stream>>>(
            scb, ada_wi, ada_bi, mods, D_MODEL, 6 * D_MODEL, 0);

        // xmod = modulate(ln(h), shift_msa, scale_msa)
        lnmod_kernel<<<MTOT, 256, 0, stream>>>(hbuf, xmod, mods, mods + D_MODEL, 6 * D_MODEL);

        // xz = xmod @ in_w + in_b                       (4096x768 . 768x3072)
        launch_gemm(xmod, in_w + (size_t)i * D_MODEL * 2 * D_INNER, in_b + (size_t)i * 2 * D_INNER,
                    xz, MTOT, 2 * D_INNER, D_MODEL, D_MODEL, 2 * D_INNER, 2 * D_INNER,
                    EP_NONE, nullptr, 0, stream);

        // u = silu(causal depthwise conv(xz[:, :D_INNER]))
        conv_silu_kernel<<<(MTOT * D_INNER + TPB - 1) / TPB, TPB, 0, stream>>>(
            xz, conv_w + (size_t)i * D_INNER * 4, conv_b + (size_t)i * D_INNER, ubuf);

        // dbl = u @ xp_w                                (4096x1536 . 1536x176)
        launch_gemm(ubuf, xp_w + (size_t)i * D_INNER * DBL_N, nullptr,
                    dblb, MTOT, DBL_N, D_INNER, D_INNER, DBL_N, DBL_N,
                    EP_NONE, nullptr, 0, stream);

        // dt = softplus(dbl[:, :48] @ dtp_w + dtp_b)    (4096x48 . 48x1536)
        launch_gemm(dblb, dtp_w + (size_t)i * DTR * D_INNER, dtp_b + (size_t)i * D_INNER,
                    dtb, MTOT, D_INNER, DTR, DBL_N, D_INNER, D_INNER,
                    EP_SOFTPLUS, nullptr, 0, stream);

        // selective scan
        scan_kernel<<<dim3(D_INNER / 256, NBATCH), 256, 0, stream>>>(
            ubuf, dtb, dblb, A_log + (size_t)i * D_INNER * NSTATE, Dp + (size_t)i * D_INNER, ybuf);

        // y *= silu(z)
        gatez_kernel<<<(MTOT * D_INNER + TPB - 1) / TPB, TPB, 0, stream>>>(xz, ybuf);

        // h += gate_msa * (y @ out_w + out_b)           (4096x1536 . 1536x768)
        launch_gemm(ybuf, out_w + (size_t)i * D_INNER * D_MODEL, out_b + (size_t)i * D_MODEL,
                    hbuf, MTOT, D_MODEL, D_INNER, D_INNER, D_MODEL, D_MODEL,
                    EP_GATE_RES, mods + 2 * D_MODEL, 6 * D_MODEL, stream);

        // MLP branch
        lnmod_kernel<<<MTOT, 256, 0, stream>>>(hbuf, xmod, mods + 3 * D_MODEL, mods + 4 * D_MODEL, 6 * D_MODEL);
        launch_gemm(xmod, fc1_w + (size_t)i * D_MODEL * 4 * D_MODEL, fc1_b + (size_t)i * 4 * D_MODEL,
                    xz, MTOT, 4 * D_MODEL, D_MODEL, D_MODEL, 4 * D_MODEL, 4 * D_MODEL,
                    EP_GELU, nullptr, 0, stream);
        launch_gemm(xz, fc2_w + (size_t)i * 4 * D_MODEL * D_MODEL, fc2_b + (size_t)i * D_MODEL,
                    hbuf, MTOT, D_MODEL, 4 * D_MODEL, 4 * D_MODEL, D_MODEL, D_MODEL,
                    EP_GATE_RES, mods + 5 * D_MODEL, 6 * D_MODEL, stream);
    }

    // final layer
    gemv_kernel<<<(NBATCH * 2 * D_MODEL + TPB - 1) / TPB, TPB, 0, stream>>>(
        scb, flaw, flab, fmods, D_MODEL, 2 * D_MODEL, 0);
    lnmod_kernel<<<MTOT, 256, 0, stream>>>(hbuf, xmod, fmods, fmods + D_MODEL, 2 * D_MODEL);
    launch_gemm(xmod, fl_w, fl_b, (float*)d_out, MTOT, 16, D_MODEL, D_MODEL, 16, 16,
                EP_NONE, nullptr, 0, stream);
}
